// MICRO_Graph_53532472378063
// MI455X (gfx1250) — matrix operand order repaired and checked by reading, not compiled, not run on hardware
//
#include <hip/hip_runtime.h>
#include <hip/hip_bf16.h>

#define NN 500000
#define DD 64
#define KK 20
#define EE 8000000
#define LAMB_ 20.0f

typedef __attribute__((ext_vector_type(16))) _Float16 v16h;
typedef __attribute__((ext_vector_type(8)))  float    v8f;

union Frag { v16h v; _Float16 h[16]; unsigned int u[8]; };

static __device__ __forceinline__ unsigned int pack2f16(float a, float b) {
    _Float16 lo = (_Float16)a, hi = (_Float16)b;
    unsigned short ul, uh;
    __builtin_memcpy(&ul, &lo, 2);
    __builtin_memcpy(&uh, &hi, 2);
    return (unsigned int)ul | ((unsigned int)uh << 16);
}

// ---- DPP-based 16-lane butterfly reductions (pure VALU, no LDS traffic) ----
// quad_perm xor1 = 0xB1, quad_perm xor2 = 0x4E, row_half_mirror = 0x141,
// row_mirror = 0x140. DPP rows (16 lanes) == our reduction groups.
template <int CTRL>
static __device__ __forceinline__ float dpp_xchg(float x) {
    return __int_as_float(__builtin_amdgcn_update_dpp(
        0, __float_as_int(x), CTRL, 0xF, 0xF, true));
}
static __device__ __forceinline__ float red16_sum(float x) {
    x += dpp_xchg<0xB1>(x);
    x += dpp_xchg<0x4E>(x);
    x += dpp_xchg<0x141>(x);
    x += dpp_xchg<0x140>(x);
    return x;
}
static __device__ __forceinline__ float red16_max(float x) {
    x = fmaxf(x, dpp_xchg<0xB1>(x));
    x = fmaxf(x, dpp_xchg<0x4E>(x));
    x = fmaxf(x, dpp_xchg<0x141>(x));
    x = fmaxf(x, dpp_xchg<0x140>(x));
    return x;
}

// ---------------- init: v=1, zero accumulators ----------------
__global__ void k_init(float* __restrict__ vv, float* __restrict__ S1,
                       float* __restrict__ s_pool, float* __restrict__ counts,
                       float* __restrict__ scal) {
    int i = blockIdx.x * blockDim.x + threadIdx.x;
    if (i < NN) vv[i] = 1.0f;
    if (i < KK) { S1[i] = 0.0f; counts[i] = 0.0f; }
    if (i < KK * DD) s_pool[i] = 0.0f;
    if (i < 4) scal[i] = 0.0f;
}

// ---------------- prep: normalize M, pack B-fragments ----------------
__global__ void k_prep(const float* __restrict__ M, const float* __restrict__ W_h,
                       float* __restrict__ Mn, unsigned int* __restrict__ Bf1,
                       unsigned int* __restrict__ Bf2) {
    __shared__ float nrm[KK];
    int t = threadIdx.x;
    if (t < KK) {
        float s = 0.0f;
        for (int d = 0; d < DD; d++) { float x = M[t * DD + d]; s += x * x; }
        nrm[t] = fmaxf(sqrtf(s), 1e-8f);
    }
    __syncthreads();
    for (int i = t; i < KK * DD; i += 256) Mn[i] = M[i] / nrm[i / DD];
    __syncthreads();
    for (int i = t; i < 4 * 2 * 8 * 32; i += 256) {
        int lane = i & 31, v = (i >> 5) & 7, c = (i >> 8) & 1, j = (i >> 9) & 3;
        int n = 16 * j + (lane & 15);
        int k = 32 * c + ((lane < 16) ? 0 : 16) + 2 * v;
        Bf1[i] = pack2f16(W_h[n * DD + k], W_h[n * DD + k + 1]);
    }
    for (int i = t; i < 2 * 2 * 8 * 32; i += 256) {
        int lane = i & 31, v = (i >> 5) & 7, c = (i >> 8) & 1, j = (i >> 9) & 1;
        int kk = 16 * j + (lane & 15);
        int d = 32 * c + ((lane < 16) ? 0 : 16) + 2 * v;
        float a = (kk < KK) ? Mn[kk * DD + d]     : 0.0f;
        float b = (kk < KK) ? Mn[kk * DD + d + 1] : 0.0f;
        Bf2[i] = pack2f16(a, b);
    }
}

// ---------------- fused WMMA: Q = softmax(cos(hW, Mn)), E = exp(L*Q) ----------------
__global__ __launch_bounds__(256) void k_q(const float* __restrict__ h,
                                           const unsigned int* __restrict__ Bf1,
                                           const unsigned int* __restrict__ Bf2,
                                           float* __restrict__ Q,
                                           float* __restrict__ Ebuf) {
    __shared__ _Float16 hb[8][16 * DD];   // per-wave 16x64 f16 transpose buffer
    const int lane = threadIdx.x & 31;
    const int wv   = threadIdx.x >> 5;
    const int gw   = blockIdx.x * 8 + wv;
    const int nw   = gridDim.x * 8;
    const int m16  = lane & 15;
    const int hi8  = (lane >= 16) ? 8 : 0;

    Frag b1[4][2], b2[2][2];
#pragma unroll
    for (int j = 0; j < 4; j++)
#pragma unroll
        for (int c = 0; c < 2; c++)
#pragma unroll
            for (int v = 0; v < 8; v++)
                b1[j][c].u[v] = Bf1[(((j * 2 + c) * 8 + v) << 5) + lane];
#pragma unroll
    for (int j = 0; j < 2; j++)
#pragma unroll
        for (int c = 0; c < 2; c++)
#pragma unroll
            for (int v = 0; v < 8; v++)
                b2[j][c].u[v] = Bf2[(((j * 2 + c) * 8 + v) << 5) + lane];

    for (int tile = gw; tile < NN / 16; tile += nw) {
        const float* hrow = h + (size_t)(tile * 16 + m16) * DD;
        Frag a0, a1;
#pragma unroll
        for (int v = 0; v < 8; v++) {
            int k = 2 * hi8 + 2 * v;
            float2 p0 = *(const float2*)(hrow + k);
            float2 p1 = *(const float2*)(hrow + k + 32);
            a0.h[2 * v] = (_Float16)p0.x; a0.h[2 * v + 1] = (_Float16)p0.y;
            a1.h[2 * v] = (_Float16)p1.x; a1.h[2 * v + 1] = (_Float16)p1.y;
        }
        v8f acc[4];
#pragma unroll
        for (int j = 0; j < 4; j++) {
            v8f c0 = {};
            c0 = __builtin_amdgcn_wmma_f32_16x16x32_f16(false, a0.v, false, b1[j][0].v,
                                                        (short)0, c0, false, false);
            c0 = __builtin_amdgcn_wmma_f32_16x16x32_f16(false, a1.v, false, b1[j][1].v,
                                                        (short)0, c0, false, false);
            acc[j] = c0;
        }
        // per-row inverse norms (row m = v + hi8, spread across the 16-lane half)
        float rn[8];
#pragma unroll
        for (int v = 0; v < 8; v++) {
            float s = acc[0][v] * acc[0][v] + acc[1][v] * acc[1][v] +
                      acc[2][v] * acc[2][v] + acc[3][v] * acc[3][v];
            s = red16_sum(s);
            rn[v] = 1.0f / fmaxf(sqrtf(s), 1e-8f);
        }
        // spill hW tile to LDS as f16 [m][n] (C-layout -> A-layout transpose)
#pragma unroll
        for (int j = 0; j < 4; j++)
#pragma unroll
            for (int v = 0; v < 8; v++)
                hb[wv][(v + hi8) * DD + 16 * j + m16] = (_Float16)acc[j][v];
        // reload as A fragments
        const unsigned int* hbu = (const unsigned int*)hb[wv];
        Frag a20, a21;
#pragma unroll
        for (int v = 0; v < 8; v++) {
            int k = 2 * hi8 + 2 * v;
            a20.u[v] = hbu[(m16 * DD + k) >> 1];
            a21.u[v] = hbu[(m16 * DD + k + 32) >> 1];
        }
        v8f acc2[2];
#pragma unroll
        for (int j = 0; j < 2; j++) {
            v8f c0 = {};
            c0 = __builtin_amdgcn_wmma_f32_16x16x32_f16(false, a20.v, false, b2[j][0].v,
                                                        (short)0, c0, false, false);
            c0 = __builtin_amdgcn_wmma_f32_16x16x32_f16(false, a21.v, false, b2[j][1].v,
                                                        (short)0, c0, false, false);
            acc2[j] = c0;
        }
        // fused softmax over K=20 (TAU=1), store Q and E = exp(LAMB*Q)
        bool v1 = (m16 < 4);
#pragma unroll
        for (int v = 0; v < 8; v++) {
            float x0 = acc2[0][v] * rn[v];                 // motif k = m16
            float x1 = acc2[1][v] * rn[v];                 // motif k = 16+m16 (valid m16<4)
            float mx = red16_max(fmaxf(x0, v1 ? x1 : -1e30f));
            float e0 = expf(x0 - mx);
            float e1 = v1 ? expf(x1 - mx) : 0.0f;
            float s = red16_sum(e0 + e1);
            float inv = 1.0f / s;
            float q0 = e0 * inv;
            int row = tile * 16 + v + hi8;
            Q[(size_t)row * KK + m16]    = q0;
            Ebuf[(size_t)row * KK + m16] = expf(LAMB_ * q0);
            if (v1) {
                float q1 = e1 * inv;
                Q[(size_t)row * KK + 16 + m16]    = q1;
                Ebuf[(size_t)row * KK + 16 + m16] = expf(LAMB_ * q1);
            }
        }
    }
}

// ---------------- sinkhorn: S1[k] = sum_n E[n,k] * v[n] ----------------
__global__ __launch_bounds__(256) void k_rowsum(const float* __restrict__ Ebuf,
                                                const float* __restrict__ vv,
                                                float* __restrict__ S1) {
    __shared__ float ls[KK];
    int t = threadIdx.x;
    if (t < KK) ls[t] = 0.0f;
    __syncthreads();
    int n = blockIdx.x * 256 + t;
    float vn = (n < NN) ? vv[n] : 0.0f;
#pragma unroll
    for (int k = 0; k < KK; k++) {
        float s = (n < NN) ? Ebuf[(size_t)n * KK + k] * vn : 0.0f;
        s = red16_sum(s);
        if ((t & 15) == 0) atomicAdd(&ls[k], s);   // both half-leaders emit
    }
    __syncthreads();
    if (t < KK) atomicAdd(&S1[t], ls[t]);
}

__global__ void k_u(float* __restrict__ S1, float* __restrict__ u) {
    int t = threadIdx.x;
    if (t < KK) {
        u[t] = (1.0f / (float)KK) / S1[t];
        S1[t] = 0.0f;   // ready for next iteration
    }
}

__global__ __launch_bounds__(256) void k_colnorm(const float* __restrict__ Ebuf,
                                                 const float* __restrict__ u,
                                                 float* __restrict__ vv) {
    __shared__ float us[KK];
    int t = threadIdx.x;
    if (t < KK) us[t] = u[t];
    __syncthreads();
    int n = blockIdx.x * 256 + t;
    if (n >= NN) return;
    float s = 0.0f;
#pragma unroll
    for (int k = 0; k < KK; k++) s += us[k] * Ebuf[(size_t)n * KK + k];
    vv[n] = (1.0f / (float)NN) / s;
}

// ---------------- assign + node-motif loss + scatter-mean pooling ----------------
__global__ __launch_bounds__(256) void k_assign(const float* __restrict__ Q,
                                                const float* __restrict__ Ebuf,
                                                const float* __restrict__ u,
                                                const float* __restrict__ h,
                                                float* __restrict__ s_pool,
                                                float* __restrict__ counts,
                                                float* __restrict__ scal) {
    __shared__ float sl[KK * DD];
    __shared__ float cl[KK];
    __shared__ float us[KK];
    __shared__ float red[16];
    int t = threadIdx.x;
    for (int i = t; i < KK * DD; i += 256) sl[i] = 0.0f;
    if (t < KK) { cl[t] = 0.0f; us[t] = u[t]; }
    if (t < 16) red[t] = 0.0f;
    __syncthreads();
    int n = blockIdx.x * 256 + t;
    float logq = 0.0f;
    if (n < NN) {
        // argmax_k u_k * E[n,k]  (monotone-equivalent to log u + LAMB*Q)
        float best = -1.0f; int ba = 0;
#pragma unroll
        for (int k = 0; k < KK; k++) {
            float sc = us[k] * Ebuf[(size_t)n * KK + k];
            if (sc > best) { best = sc; ba = k; }
        }
        logq = logf(Q[(size_t)n * KK + ba]);
        atomicAdd(&cl[ba], 1.0f);
        const float* hr = h + (size_t)n * DD;
#pragma unroll
        for (int d = 0; d < DD; d++) atomicAdd(&sl[ba * DD + d], hr[d]);
    }
    float s = red16_sum(logq);
    if ((t & 15) == 0) red[t >> 4] = s;
    __syncthreads();
    if (t == 0) {
        float tot = 0.0f;
        for (int w = 0; w < 16; w++) tot += red[w];
        atomicAdd(&scal[0], tot);
    }
    for (int i = t; i < KK * DD; i += 256) atomicAdd(&s_pool[i], sl[i]);
    if (t < KK) atomicAdd(&counts[t], cl[t]);
}

// ---------------- spectral regularizer over edge list (Q L2-resident) ----------------
__global__ __launch_bounds__(256) void k_edges(const float* __restrict__ Q,
                                               const int* __restrict__ es,
                                               const int* __restrict__ ed,
                                               float* __restrict__ scal) {
    __shared__ float red[16];
    int t = threadIdx.x;
    if (t < 16) red[t] = 0.0f;
    float acc = 0.0f;
    for (int e = blockIdx.x * 256 + t; e < EE; e += gridDim.x * 256) {
        const float4* qa = (const float4*)(Q + (size_t)es[e] * KK);
        const float4* qb = (const float4*)(Q + (size_t)ed[e] * KK);
#pragma unroll
        for (int j = 0; j < 5; j++) {
            float4 x = qa[j], y = qb[j];
            acc += x.x * y.x + x.y * y.y + x.z * y.z + x.w * y.w;
        }
    }
    acc = red16_sum(acc);
    __syncthreads();
    if ((t & 15) == 0) red[t >> 4] = acc;
    __syncthreads();
    if (t == 0) {
        float tot = 0.0f;
        for (int w = 0; w < 16; w++) tot += red[w];
        atomicAdd(&scal[1], tot);
    }
}

// ---------------- tiny tail: motif-subgraph loss + combine ----------------
__global__ __launch_bounds__(256) void k_final(const float* __restrict__ s_pool,
                                               const float* __restrict__ counts,
                                               const float* __restrict__ W_s,
                                               const float* __restrict__ Mn,
                                               const float* __restrict__ scal,
                                               float* __restrict__ out) {
    __shared__ float sm[KK * DD];
    __shared__ float tt[KK * DD];
    __shared__ float lg[KK * KK];
    __shared__ float tn[KK];
    int t = threadIdx.x;
    for (int i = t; i < KK * DD; i += 256)
        sm[i] = s_pool[i] / fmaxf(counts[i / DD], 1.0f);
    __syncthreads();
    for (int i = t; i < KK * DD; i += 256) {
        int k = i / DD, j = i % DD;
        float a = 0.0f;
        for (int d = 0; d < DD; d++) a += sm[k * DD + d] * W_s[j * DD + d];
        tt[i] = a;
    }
    __syncthreads();
    if (t < KK) {
        float s = 0.0f;
        for (int d = 0; d < DD; d++) { float x = tt[t * DD + d]; s += x * x; }
        tn[t] = fmaxf(sqrtf(s), 1e-8f);
    }
    __syncthreads();
    for (int i = t; i < KK * KK; i += 256) {
        int k = i / KK, j = i % KK;
        float a = 0.0f;
        for (int d = 0; d < DD; d++) a += tt[k * DD + d] * Mn[j * DD + d];
        lg[i] = a / tn[k];
    }
    __syncthreads();
    if (t == 0) {
        float lm = 0.0f;
        for (int k = 0; k < KK; k++) {
            float mx = -1e30f;
            for (int j = 0; j < KK; j++) mx = fmaxf(mx, lg[k * KK + j]);
            float s = 0.0f;
            for (int j = 0; j < KK; j++) s += expf(lg[k * KK + j] - mx);
            lm += lg[k * KK + k] - (mx + logf(s));
        }
        float loss_mot_sub = -lm / (float)KK;
        float loss_node    = -scal[0] / (float)NN;
        float loss_reg     = -scal[1] / (float)EE;
        out[0] = loss_mot_sub + loss_node + loss_reg;   // ALPHA=1 drops loss_c
    }
}

extern "C" void kernel_launch(void* const* d_in, const int* in_sizes, int n_in,
                              void* d_out, int out_size, void* d_ws, size_t ws_size,
                              hipStream_t stream) {
    const float* h   = (const float*)d_in[0];
    const float* M   = (const float*)d_in[2];
    const float* W_h = (const float*)d_in[3];
    const float* W_s = (const float*)d_in[4];
    const int*   es  = (const int*)d_in[6];
    const int*   ed  = (const int*)d_in[7];
    float* out = (float*)d_out;

    float* Q      = (float*)d_ws;                 // N*K
    float* Ebuf   = Q + (size_t)NN * KK;          // N*K  (exp(LAMB*Q), L2-resident)
    float* vv     = Ebuf + (size_t)NN * KK;       // N
    float* u      = vv + NN;                      // K
    float* S1     = u + KK;                       // K
    float* s_pool = S1 + KK;                      // K*D
    float* counts = s_pool + KK * DD;             // K
    float* scal   = counts + KK;                  // 8
    float* Mn     = scal + 8;                     // K*D
    unsigned int* Bf1 = (unsigned int*)(Mn + KK * DD);  // 2048
    unsigned int* Bf2 = Bf1 + 2048;                      // 1024

    k_init<<<(NN + 255) / 256, 256, 0, stream>>>(vv, S1, s_pool, counts, scal);
    k_prep<<<1, 256, 0, stream>>>(M, W_h, Mn, Bf1, Bf2);
    k_q<<<512, 256, 0, stream>>>(h, Bf1, Bf2, Q, Ebuf);
    for (int it = 0; it < 5; it++) {
        k_rowsum<<<(NN + 255) / 256, 256, 0, stream>>>(Ebuf, vv, S1);
        k_u<<<1, 32, 0, stream>>>(S1, u);
        k_colnorm<<<(NN + 255) / 256, 256, 0, stream>>>(Ebuf, u, vv);
    }
    k_assign<<<(NN + 255) / 256, 256, 0, stream>>>(Q, Ebuf, u, h, s_pool, counts, scal);
    k_edges<<<4096, 256, 0, stream>>>(Q, es, ed, scal);
    k_final<<<1, 256, 0, stream>>>(s_pool, counts, W_s, Mn, scal, out);
}
